// SwinTransformer3DJoint_36318243455757
// MI455X (gfx1250) — matrix operand order repaired and checked by reading, not compiled, not run on hardware
//
#include <hip/hip_runtime.h>
#include <hip/hip_bf16.h>

// ---------------------------------------------------------------------------
// Swin-3D block (window attention + MLP) for gfx1250 / MI455X.
// Memory-bound (~22 GFLOP over ~170 MB @ 23.3 TB/s => ~7-8 us floor), so all
// GEMMs run in bf16 via v_wmma_f32_16x16x32_bf16 with fp32 accumulation;
// LayerNorm / softmax / GELU / residuals stay fp32.  Each GEMM wave computes
// a 16Mx64N strip (4 accumulators) so every A fragment feeds 4 WMMAs.
// ---------------------------------------------------------------------------

typedef __bf16 bf16;
typedef __attribute__((ext_vector_type(16))) __bf16 v16bf;
typedef __attribute__((ext_vector_type(8)))  float  v8f;

#define TOK     50176        // 2*8*56*56 tokens ( = 512 windows * 98 )
#define NWIN    512
#define NWH     2048         // windows * heads
#define NSEQ    98           // tokens per window
#define NPAD    128          // padded seq for K-dim of P@V
#define MTILES  3136         // TOK / 16

// ---------------- WMMA fragment helpers (ISA 7.12.2 layouts) ----------------

__device__ __forceinline__ v8f wmma_bf16(v16bf a, v16bf b, v8f c) {
  return __builtin_amdgcn_wmma_f32_16x16x32_bf16(false, a, false, b,
                                                 (short)0, c, false, false);
}

// A 16x32 bf16 fragment: lane holds row (lane&15); per-lane K chunks at
// +half*8 (elems 0..7) and +16+half*8 (elems 8..15).  `p` = row ptr + half*8.
__device__ __forceinline__ v16bf ld_fragA(const bf16* p) {
  union { v16bf v; float4 f[2]; } u;
  u.f[0] = *(const float4*)(p);
  u.f[1] = *(const float4*)(p + 16);
  return u.v;
}

// B 32x16 bf16 fragment: lane holds column (lane&15); per-lane K chunks at
// +half*8 (elems 0..7) and +16+half*8 (elems 8..15).  `p` = col-row ptr + half*8.
__device__ __forceinline__ v16bf ld_fragB(const bf16* p) {
  union { v16bf v; float4 f[2]; } u;
  u.f[0] = *(const float4*)(p);
  u.f[1] = *(const float4*)(p + 16);
  return u.v;
}

// A fragment sourced from fp32 LDS (softmax probabilities) with convert.
__device__ __forceinline__ v16bf ld_fragA_f32(const float* p) {
  float4 c0 = *(const float4*)(p);
  float4 c1 = *(const float4*)(p + 4);
  float4 c2 = *(const float4*)(p + 16);
  float4 c3 = *(const float4*)(p + 20);
  v16bf v;
  v[0]=(bf16)c0.x;  v[1]=(bf16)c0.y;  v[2]=(bf16)c0.z;  v[3]=(bf16)c0.w;
  v[4]=(bf16)c1.x;  v[5]=(bf16)c1.y;  v[6]=(bf16)c1.z;  v[7]=(bf16)c1.w;
  v[8]=(bf16)c2.x;  v[9]=(bf16)c2.y;  v[10]=(bf16)c2.z; v[11]=(bf16)c2.w;
  v[12]=(bf16)c3.x; v[13]=(bf16)c3.y; v[14]=(bf16)c3.z; v[15]=(bf16)c3.w;
  return v;
}

// 16M x 64N strip per wave: A row-major (stride K), B row-major (N_out x K).
// One A fragment feeds 4 WMMAs per K chunk.
template<int K>
__device__ __forceinline__ void gemm_strip4(const bf16* A, const bf16* B,
                                            v8f acc[4]) {
  const int lane = threadIdx.x & 31;
  const int hf   = lane >> 4;
  const int idx  = lane & 15;
  const bf16* ap = A + idx * K + hf * 8;
  const bf16* bp = B + idx * K + hf * 8;
#pragma unroll
  for (int k0 = 0; k0 < K; k0 += 32) {
    v16bf a = ld_fragA(ap + k0);
#pragma unroll
    for (int g = 0; g < 4; ++g)
      acc[g] = wmma_bf16(a, ld_fragB(bp + (size_t)g * 16 * K + k0), acc[g]);
  }
}

__device__ __forceinline__ float wred(float v) {
#pragma unroll
  for (int m = 16; m > 0; m >>= 1) v += __shfl_xor(v, m, 32);
  return v;
}

// ---------------- small prep kernels ----------------

__global__ void k_cvt(const float* __restrict__ s, bf16* __restrict__ d, int n) {
  int i = blockIdx.x * 256 + threadIdx.x;
  if (i < n) d[i] = (bf16)s[i];
}

// bias_buf[h][q][k] = bias_table[rel_index[q*98+k]][h]
__global__ void k_bias(const float* __restrict__ table, const int* __restrict__ rel,
                       float* __restrict__ out) {
  int i = blockIdx.x * 256 + threadIdx.x;
  if (i < NSEQ * NSEQ) {
    int ri = rel[i];
#pragma unroll
    for (int h = 0; h < 4; ++h) out[h * NSEQ * NSEQ + i] = table[ri * 4 + h];
  }
}

// ---------------- LayerNorm (one wave32 per token) ----------------
// permute==1: write into window-partitioned layout (win*98+n).
__global__ __launch_bounds__(256)
void k_ln(const float* __restrict__ x, const float* __restrict__ g,
          const float* __restrict__ b, bf16* __restrict__ dst, int permute) {
  int wave = threadIdx.x >> 5, lane = threadIdx.x & 31;
  long t = (long)blockIdx.x * 8 + wave;            // token id, 0..50175
  const float* row = x + t * 128;
  float4 v = *(const float4*)(row + lane * 4);
  float mean = wred(v.x + v.y + v.z + v.w) * (1.f / 128.f);
  float4 d2 = make_float4(v.x - mean, v.y - mean, v.z - mean, v.w - mean);
  float var = wred(d2.x*d2.x + d2.y*d2.y + d2.z*d2.z + d2.w*d2.w) * (1.f / 128.f);
  float rstd = rsqrtf(var + 1e-5f);

  long r;
  if (permute) {
    int wq = t % 56, hq = (t / 56) % 56, dq = (t / 3136) % 8, bq = t / 25088;
    int win = ((bq * 4 + (dq >> 1)) * 8 + hq / 7) * 8 + wq / 7;
    int n   = (dq & 1) * 49 + (hq % 7) * 7 + (wq % 7);
    r = (long)win * NSEQ + n;
  } else {
    r = t;
  }
  int c = lane * 4;
  bf16* o = dst + r * 128 + c;
  o[0] = (bf16)(d2.x * rstd * g[c]     + b[c]);
  o[1] = (bf16)(d2.y * rstd * g[c + 1] + b[c + 1]);
  o[2] = (bf16)(d2.z * rstd * g[c + 2] + b[c + 2]);
  o[3] = (bf16)(d2.w * rstd * g[c + 3] + b[c + 3]);
}

// ---------------- QKV GEMM: (50176x128)bf16 @ (384x128)^T ----------------
__global__ __launch_bounds__(256)
void k_qkv(const bf16* __restrict__ A, const bf16* __restrict__ W,
           const float* __restrict__ qb_bias,
           bf16* __restrict__ qb, bf16* __restrict__ kb, bf16* __restrict__ vtb) {
  int wave = threadIdx.x >> 5;
  int strip = blockIdx.x * 8 + wave;
  const int NG = 6;                        // 384 / 64
  int mt = strip / NG, ng = strip % NG;
  if (mt >= MTILES) return;
  v8f acc[4] = {};
  gemm_strip4<128>(A + (size_t)mt * 16 * 128, W + (size_t)ng * 64 * 128, acc);

  int lane = threadIdx.x & 31, hf = lane >> 4, cn = lane & 15;
#pragma unroll
  for (int g = 0; g < 4; ++g) {
    int j = ng * 64 + g * 16 + cn;         // 0..383
    int sec = j >> 7, head = (j >> 5) & 3, dd = j & 31;
    float bias = qb_bias[j];
#pragma unroll
    for (int rr = 0; rr < 8; ++rr) {
      int m = mt * 16 + rr + hf * 8;
      int win = m / NSEQ, n = m % NSEQ;
      float v = acc[g][rr] + bias;
      size_t wh = (size_t)win * 4 + head;
      if (sec == 0) {                      // q, pre-scaled by hd^-0.5
        v *= 0.17677669529663687f;
        qb[(wh * NPAD + n) * 32 + dd] = (bf16)v;
      } else if (sec == 1) {
        kb[(wh * NPAD + n) * 32 + dd] = (bf16)v;
      } else {                             // v stored transposed: (wh, d, n)
        vtb[(wh * 32 + dd) * NPAD + n] = (bf16)v;
      }
    }
  }
}

// ---------------- attention: one workgroup per (window, head) ----------------
__global__ __launch_bounds__(256)
void k_attn(const bf16* __restrict__ qb, const bf16* __restrict__ kb,
            const bf16* __restrict__ vtb, const float* __restrict__ bias,
            bf16* __restrict__ attn_out) {
  __shared__ float sc[112 * 128];          // 57,344 B of the 320 KB WGP LDS

  int wh = blockIdx.x;                     // 0..2047
  int win = wh >> 2, head = wh & 3;
  const bf16* q  = qb  + (size_t)wh * NPAD * 32;
  const bf16* k  = kb  + (size_t)wh * NPAD * 32;
  const bf16* vt = vtb + (size_t)wh * 32 * NPAD;
  const float* bi = bias + head * NSEQ * NSEQ;

  int wave = threadIdx.x >> 5, lane = threadIdx.x & 31;
  int hf = lane >> 4, idx = lane & 15;

  // ---- S = q @ k^T + bias : 7 x 8 tiles of 16x16, K = 32 (single WMMA) ----
  {
    int nt = wave;                         // 8 waves == 8 N tiles
    const bf16* bp = k + (nt * 16 + idx) * 32 + hf * 8;
    v16bf bfrag = ld_fragB(bp);
    for (int mt = 0; mt < 7; ++mt) {
      const bf16* ap = q + (mt * 16 + idx) * 32 + hf * 8;
      v8f acc = {};
      acc = wmma_bf16(ld_fragA(ap), bfrag, acc);
#pragma unroll
      for (int rr = 0; rr < 8; ++rr) {
        int qi = mt * 16 + rr + hf * 8;    // query row 0..111
        int ki = nt * 16 + idx;            // key col   0..127
        float v;
        if (qi < NSEQ && ki < NSEQ) v = acc[rr] + bi[qi * NSEQ + ki];
        else                        v = (qi < NSEQ) ? -1e30f : 0.f;
        sc[qi * 128 + ki] = v;
      }
    }
  }
  __syncthreads();

  // ---- softmax, one row per thread (fp32, in place) ----
  {
    int t = threadIdx.x;
    if (t < NSEQ) {
      float* r = sc + t * 128;
      float mx = -1e30f;
      for (int m = 0; m < NSEQ; ++m) mx = fmaxf(mx, r[m]);
      float s = 0.f;
      for (int m = 0; m < NSEQ; ++m) { float e = __expf(r[m] - mx); r[m] = e; s += e; }
      float inv = 1.f / s;
      for (int m = 0; m < NSEQ; ++m) r[m] *= inv;
      for (int m = NSEQ; m < 128; ++m) r[m] = 0.f;
    } else if (t < 112) {
      float* r = sc + t * 128;
      for (int m = 0; m < 128; ++m) r[m] = 0.f;
    }
  }
  __syncthreads();

  // ---- O = P @ V : 7 x 2 tiles, K = 128 in 4 chunks ----
  for (int tile = wave; tile < 14; tile += 8) {
    int mt = tile >> 1, nt = tile & 1;
    const float* ap = sc + (mt * 16 + idx) * 128 + hf * 8;
    const bf16*  bp = vt + (nt * 16 + idx) * 128 + hf * 8;
    v8f acc = {};
#pragma unroll
    for (int k0 = 0; k0 < 128; k0 += 32)
      acc = wmma_bf16(ld_fragA_f32(ap + k0), ld_fragB(bp + k0), acc);
#pragma unroll
    for (int rr = 0; rr < 8; ++rr) {
      int qi = mt * 16 + rr + hf * 8;
      if (qi < NSEQ) {
        int dd = nt * 16 + idx;
        attn_out[((size_t)win * NSEQ + qi) * 128 + head * 32 + dd] = (bf16)acc[rr];
      }
    }
  }
}

// ---------------- proj GEMM + window reverse + residual ----------------
__global__ __launch_bounds__(256)
void k_proj(const bf16* __restrict__ A, const bf16* __restrict__ W,
            const float* __restrict__ pb, const float* __restrict__ x,
            float* __restrict__ xres) {
  int wave = threadIdx.x >> 5;
  int strip = blockIdx.x * 8 + wave;
  const int NG = 2;                        // 128 / 64
  int mt = strip / NG, ng = strip % NG;
  if (mt >= MTILES) return;
  v8f acc[4] = {};
  gemm_strip4<128>(A + (size_t)mt * 16 * 128, W + (size_t)ng * 64 * 128, acc);

  int lane = threadIdx.x & 31, hf = lane >> 4, cn = lane & 15;
#pragma unroll
  for (int g = 0; g < 4; ++g) {
    int j = ng * 64 + g * 16 + cn;
    float bias = pb[j];
#pragma unroll
    for (int rr = 0; rr < 8; ++rr) {
      int m = mt * 16 + rr + hf * 8;
      int win = m / NSEQ, n = m % NSEQ;
      int w7 = win & 7, h7 = (win >> 3) & 7, d2 = (win >> 6) & 3, b = win >> 8;
      int dn = n / 49, rem = n % 49, hn = rem / 7, wn = rem % 7;
      int d = d2 * 2 + dn, hh = h7 * 7 + hn, ww = w7 * 7 + wn;
      size_t t = (((size_t)b * 8 + d) * 56 + hh) * 56 + ww;
      xres[t * 128 + j] = x[t * 128 + j] + acc[g][rr] + bias;
    }
  }
}

// ---------------- MLP fc1 (GELU) and fc2 (+residual) ----------------
__global__ __launch_bounds__(256)
void k_fc1(const bf16* __restrict__ A, const bf16* __restrict__ W,
           const float* __restrict__ fb, bf16* __restrict__ g_out) {
  int wave = threadIdx.x >> 5;
  int strip = blockIdx.x * 8 + wave;
  const int NG = 8;                        // 512 / 64
  int mt = strip / NG, ng = strip % NG;
  if (mt >= MTILES) return;
  v8f acc[4] = {};
  gemm_strip4<128>(A + (size_t)mt * 16 * 128, W + (size_t)ng * 64 * 128, acc);

  int lane = threadIdx.x & 31, hf = lane >> 4, cn = lane & 15;
#pragma unroll
  for (int g = 0; g < 4; ++g) {
    int j = ng * 64 + g * 16 + cn;
    float bias = fb[j];
#pragma unroll
    for (int rr = 0; rr < 8; ++rr) {
      int m = mt * 16 + rr + hf * 8;
      float v = acc[g][rr] + bias;
      v = 0.5f * v * (1.f + erff(v * 0.70710678118654752f));   // exact GELU
      g_out[(size_t)m * 512 + j] = (bf16)v;
    }
  }
}

__global__ __launch_bounds__(256)
void k_fc2(const bf16* __restrict__ A, const bf16* __restrict__ W,
           const float* __restrict__ fb, const float* __restrict__ xres,
           float* __restrict__ out) {
  int wave = threadIdx.x >> 5;
  int strip = blockIdx.x * 8 + wave;
  const int NG = 2;                        // 128 / 64
  int mt = strip / NG, ng = strip % NG;
  if (mt >= MTILES) return;
  v8f acc[4] = {};
  gemm_strip4<512>(A + (size_t)mt * 16 * 512, W + (size_t)ng * 64 * 512, acc);

  int lane = threadIdx.x & 31, hf = lane >> 4, cn = lane & 15;
#pragma unroll
  for (int g = 0; g < 4; ++g) {
    int j = ng * 64 + g * 16 + cn;
    float bias = fb[j];
#pragma unroll
    for (int rr = 0; rr < 8; ++rr) {
      int m = mt * 16 + rr + hf * 8;
      out[(size_t)m * 128 + j] = xres[(size_t)m * 128 + j] + acc[g][rr] + bias;
    }
  }
}

// ---------------------------------------------------------------------------

extern "C" void kernel_launch(void* const* d_in, const int* in_sizes, int n_in,
                              void* d_out, int out_size, void* d_ws, size_t ws_size,
                              hipStream_t stream) {
  const float* x        = (const float*)d_in[0];
  const float* norm1_w  = (const float*)d_in[1];
  const float* norm1_b  = (const float*)d_in[2];
  const float* qkv_w    = (const float*)d_in[3];
  const float* qkv_b    = (const float*)d_in[4];
  const float* btable   = (const float*)d_in[5];
  const float* proj_w   = (const float*)d_in[6];
  const float* proj_b   = (const float*)d_in[7];
  const float* norm2_w  = (const float*)d_in[8];
  const float* norm2_b  = (const float*)d_in[9];
  const float* fc1_w    = (const float*)d_in[10];
  const float* fc1_b    = (const float*)d_in[11];
  const float* fc2_w    = (const float*)d_in[12];
  const float* fc2_b    = (const float*)d_in[13];
  const int*   rel_idx  = (const int*)d_in[14];
  float* out = (float*)d_out;

  // workspace carve (all 256-B aligned); ~166 MB total
  char* ws = (char*)d_ws;
  size_t off = 0;
  auto alloc = [&](size_t bytes) -> void* {
    off = (off + 255) & ~(size_t)255;
    void* p = ws + off; off += bytes; return p;
  };
  bf16*  xw    = (bf16*)alloc((size_t)TOK * 128 * 2);
  bf16*  qb    = (bf16*)alloc((size_t)NWH * NPAD * 32 * 2);
  bf16*  kb    = (bf16*)alloc((size_t)NWH * NPAD * 32 * 2);
  bf16*  vtb   = (bf16*)alloc((size_t)NWH * 32 * NPAD * 2);
  bf16*  attn  = (bf16*)alloc((size_t)TOK * 128 * 2);
  float* xres  = (float*)alloc((size_t)TOK * 128 * 4);
  bf16*  hbuf  = (bf16*)alloc((size_t)TOK * 128 * 2);
  bf16*  gbuf  = (bf16*)alloc((size_t)TOK * 512 * 2);
  bf16*  wqkv  = (bf16*)alloc((size_t)384 * 128 * 2);
  bf16*  wproj = (bf16*)alloc((size_t)128 * 128 * 2);
  bf16*  wfc1  = (bf16*)alloc((size_t)512 * 128 * 2);
  bf16*  wfc2  = (bf16*)alloc((size_t)128 * 512 * 2);
  float* bbuf  = (float*)alloc((size_t)4 * NSEQ * NSEQ * 4);

  // weight conversions + bias gather
  k_cvt<<<(49152 + 255) / 256, 256, 0, stream>>>(qkv_w,  wqkv,  49152);
  k_cvt<<<(16384 + 255) / 256, 256, 0, stream>>>(proj_w, wproj, 16384);
  k_cvt<<<(65536 + 255) / 256, 256, 0, stream>>>(fc1_w,  wfc1,  65536);
  k_cvt<<<(65536 + 255) / 256, 256, 0, stream>>>(fc2_w,  wfc2,  65536);
  k_bias<<<(NSEQ * NSEQ + 255) / 256, 256, 0, stream>>>(btable, rel_idx, bbuf);

  // zero the seq-padding rows of q/k/v (98 -> 128)
  hipMemsetAsync(qb,  0, (size_t)NWH * NPAD * 32 * 2, stream);
  hipMemsetAsync(kb,  0, (size_t)NWH * NPAD * 32 * 2, stream);
  hipMemsetAsync(vtb, 0, (size_t)NWH * 32 * NPAD * 2, stream);

  // 1) LN1 + window partition -> bf16
  k_ln<<<TOK / 8, 256, 0, stream>>>(x, norm1_w, norm1_b, xw, 1);
  // 2) QKV GEMM: 3136 Mtiles x 6 N-strips, 8 waves/block
  k_qkv<<<MTILES * 6 / 8, 256, 0, stream>>>(xw, wqkv, qkv_b, qb, kb, vtb);
  // 3) window attention, one WG per (window, head)
  k_attn<<<NWH, 256, 0, stream>>>(qb, kb, vtb, bbuf, attn);
  // 4) proj + window reverse + residual: 3136 x 2 strips
  k_proj<<<MTILES * 2 / 8, 256, 0, stream>>>(attn, wproj, proj_b, x, xres);
  // 5) LN2 -> bf16
  k_ln<<<TOK / 8, 256, 0, stream>>>(xres, norm2_w, norm2_b, hbuf, 0);
  // 6) fc1 + GELU: 3136 x 8 strips
  k_fc1<<<MTILES * 8 / 8, 256, 0, stream>>>(hbuf, wfc1, fc1_b, gbuf);
  // 7) fc2 + residual -> output: 3136 x 2 strips
  k_fc2<<<MTILES * 2 / 8, 256, 0, stream>>>(gbuf, wfc2, fc2_b, xres, out);
}
